// Block_11003706212352
// MI455X (gfx1250) — compile-verified, matrix operand order checked by reading, not run on hardware
//
#include <hip/hip_runtime.h>
#include <hip/hip_bf16.h>
#include <math.h>

// ---------------------------------------------------------------------------
// RWKV-v5 block forward for MI455X (gfx1250).
// GEMMs: bf16 WMMA (v_wmma_f32_16x16x32_bf16) with double-buffered LDS stages
// filled by gfx1250 async copies (global_load_async_to_lds_b128 / ASYNCcnt).
// WKV5 scan: register-resident 64x64 state per (b,h), LDS broadcast.
// ---------------------------------------------------------------------------

using bf16 = __bf16;
typedef __attribute__((ext_vector_type(16))) __bf16 bf16x16;
typedef __attribute__((ext_vector_type(8)))  __bf16 bf16x8;
typedef __attribute__((ext_vector_type(8)))  float  v8f;

constexpr int   Bc  = 2;
constexpr int   Tc  = 1024;
constexpr int   Cc  = 2048;
constexpr int   Hc  = 32;
constexpr int   FFc = 8192;
constexpr int   BT  = Bc * Tc;                 // 2048 rows
constexpr size_t BTC = (size_t)BT * Cc;        // 4M elements
constexpr float EPSf = 1e-5f;
constexpr float INV_DIV = 0.125f;              // 1/8.0

// LDS staging geometry for the GEMM: 128-row slab, 64 data bytes (32 bf16 K)
// per row, padded to 80B (20-bank stride) to spread ds_load_b128 accesses.
constexpr int ROW_STRIDE  = 80;
constexpr int SLAB_BYTES  = 128 * ROW_STRIDE;  // 10240
constexpr int STAGE_BYTES = 2 * SLAB_BYTES;    // A + B = 20480

#define SHUF16(lo, hi) __builtin_shufflevector((lo), (hi), 0,1,2,3,4,5,6,7,8,9,10,11,12,13,14,15)

__device__ __forceinline__ v8f wmma_bf16(bf16x16 a, bf16x16 b, v8f c) {
  return __builtin_amdgcn_wmma_f32_16x16x32_bf16(false, a, false, b, (short)0, c,
                                                 false, false);
}

// Async 16B global -> LDS copy (gfx1250). VDST = LDS byte address (low 32 bits
// of the generic shared pointer), VADDR = 64-bit global address, SADDR = off.
__device__ __forceinline__ void async_copy_b128(const void* gptr, void* lptr) {
  const unsigned long long ga = (unsigned long long)(uintptr_t)gptr;
  const unsigned la = (unsigned)(uintptr_t)lptr;
  asm volatile("global_load_async_to_lds_b128 %0, %1, off"
               :: "v"(la), "v"(ga) : "memory");
}

// ---------------------------------------------------------------------------
// Weight convert + transpose: W[K][N] fp32 -> Wt[N][K] bf16 (LDS-tiled)
// ---------------------------------------------------------------------------
__global__ void transpose_to_bf16(const float* __restrict__ in, bf16* __restrict__ out,
                                  int K, int N) {
  __shared__ float tile[32][33];
  const int kb = blockIdx.y * 32;
  const int nb = blockIdx.x * 32;
  const int tx = threadIdx.x;   // 0..31
  const int ty = threadIdx.y;   // 0..7
  #pragma unroll
  for (int i = ty; i < 32; i += 8)
    tile[i][tx] = in[(size_t)(kb + i) * N + nb + tx];
  __syncthreads();
  #pragma unroll
  for (int i = ty; i < 32; i += 8)
    out[(size_t)(nb + i) * K + kb + tx] = (bf16)tile[tx][i];
}

// ---------------------------------------------------------------------------
// LayerNorm: one block (256 threads) per row of C
// ---------------------------------------------------------------------------
__global__ void layernorm_kernel(const float* __restrict__ x, const float* __restrict__ g,
                                 const float* __restrict__ b, float* __restrict__ out, int C) {
  __shared__ float s1[256], s2[256];
  const int tid = threadIdx.x;
  const size_t base = (size_t)blockIdx.x * C;
  float sum = 0.f, sum2 = 0.f;
  for (int c = tid; c < C; c += 256) {
    float v = x[base + c];
    sum += v; sum2 += v * v;
  }
  s1[tid] = sum; s2[tid] = sum2;
  __syncthreads();
  for (int off = 128; off > 0; off >>= 1) {
    if (tid < off) { s1[tid] += s1[tid + off]; s2[tid] += s2[tid + off]; }
    __syncthreads();
  }
  const float mean = s1[0] / (float)C;
  const float var  = s2[0] / (float)C - mean * mean;
  const float rstd = rsqrtf(var + EPSf);
  for (int c = tid; c < C; c += 256)
    out[base + c] = (x[base + c] - mean) * rstd * g[c] + b[c];
}

// ---------------------------------------------------------------------------
// Time-shift mixes
// ---------------------------------------------------------------------------
__global__ void mix4_kernel(const float* __restrict__ xl,
                            const float* __restrict__ tmk, const float* __restrict__ tmv,
                            const float* __restrict__ tmr, const float* __restrict__ tmg,
                            bf16* __restrict__ ok, bf16* __restrict__ ov,
                            bf16* __restrict__ orr, bf16* __restrict__ og,
                            int T, int C) {
  const size_t idx = (size_t)blockIdx.x * 256 + threadIdx.x;
  if (idx >= BTC) return;
  const int c = (int)(idx % C);
  const size_t bt = idx / C;
  const int t = (int)(bt % T);
  const float cur  = xl[idx];
  const float prev = (t > 0) ? xl[idx - C] : 0.f;
  const float mk = tmk[c], mv = tmv[c], mr = tmr[c], mg = tmg[c];
  ok [idx] = (bf16)(cur * mk + prev * (1.f - mk));
  ov [idx] = (bf16)(cur * mv + prev * (1.f - mv));
  orr[idx] = (bf16)(cur * mr + prev * (1.f - mr));
  og [idx] = (bf16)(cur * mg + prev * (1.f - mg));
}

__global__ void mix2_kernel(const float* __restrict__ xl,
                            const float* __restrict__ fmk, const float* __restrict__ fmr,
                            bf16* __restrict__ ck, bf16* __restrict__ cr,
                            int T, int C) {
  const size_t idx = (size_t)blockIdx.x * 256 + threadIdx.x;
  if (idx >= BTC) return;
  const int c = (int)(idx % C);
  const size_t bt = idx / C;
  const int t = (int)(bt % T);
  const float cur  = xl[idx];
  const float prev = (t > 0) ? xl[idx - C] : 0.f;
  const float mk = fmk[c], mr = fmr[c];
  ck[idx] = (bf16)(cur * mk + prev * (1.f - mk));
  cr[idx] = (bf16)(cur * mr + prev * (1.f - mr));
}

// ---------------------------------------------------------------------------
// bf16 WMMA GEMM with async-LDS double buffering.
//   out[M,N] = epilogue( A[M,K] @ Bt[N,K]^T )
// Block: 256 threads / 8 waves; block tile 128(M) x 128(N); wave tile 32x64.
// Per K-step (32): each thread issues 4 async b128 copies (A slab + B slab);
// in-order ASYNCcnt completion lets us wait <=4 while next stage is in flight.
// Epilogue modes:
//   0: outF = acc                    1: outF = silu(acc)
//   2: outF = acc + res              3: outB = relu(acc)^2  (bf16)
//   4: outF = sigmoid(acc)           5: outF += scale * acc
// ---------------------------------------------------------------------------
__device__ __forceinline__ void gemm_issue_stage(const bf16* gA, const bf16* gB, int K,
                                                 char* ldsA, char* ldsB, int tid) {
  // 128 rows x 64B per slab = 512 x 16B chunks; 256 threads x 2 chunks each.
  #pragma unroll
  for (int c = 0; c < 2; ++c) {
    const int chunk = tid + 256 * c;
    const int row = chunk >> 2;
    const int sub = chunk & 3;
    async_copy_b128(gA + (size_t)row * K + sub * 8, ldsA + row * ROW_STRIDE + sub * 16);
    async_copy_b128(gB + (size_t)row * K + sub * 8, ldsB + row * ROW_STRIDE + sub * 16);
  }
}

__global__ void __launch_bounds__(256)
gemm_bf16(const bf16* __restrict__ A, const bf16* __restrict__ Bt,
          float* __restrict__ outF, bf16* __restrict__ outB,
          const float* __restrict__ res, const float* __restrict__ scale,
          int M, int N, int K, int mode) {
  __shared__ char smem[2 * STAGE_BYTES];       // 40 KB: 2 stages x (A slab + B slab)

  const int tid  = threadIdx.x;
  const int lane = tid & 31;
  const int wave = tid >> 5;
  const int wm   = wave & 3;                   // 4 waves along M (32 rows each)
  const int wn   = wave >> 2;                  // 2 waves along N (64 cols each)
  const int kh   = lane >> 4;                  // K-half select (0/1)
  const int l15  = lane & 15;
  const int bm0  = blockIdx.y * 128;
  const int bn0  = blockIdx.x * 128;

  v8f acc[2][4];
  #pragma unroll
  for (int a = 0; a < 2; ++a)
    #pragma unroll
    for (int j = 0; j < 4; ++j)
      #pragma unroll
      for (int e = 0; e < 8; ++e) acc[a][j][e] = 0.f;

  const bf16* gA = A  + (size_t)bm0 * K;
  const bf16* gB = Bt + (size_t)bn0 * K;

  // prologue: stage 0 into first half of smem
  gemm_issue_stage(gA, gB, K, smem, smem + SLAB_BYTES, tid);

  const int nk = K >> 5;
  for (int i = 0; i < nk; ++i) {
    if (i + 1 < nk) {
      char* nb = smem + ((i + 1) & 1) * STAGE_BYTES;     // runtime-selected stage
      gemm_issue_stage(gA + (size_t)(i + 1) * 32, gB + (size_t)(i + 1) * 32, K,
                       nb, nb + SLAB_BYTES, tid);
      asm volatile("s_wait_asynccnt 0x4" ::: "memory");  // stage i landed
    } else {
      asm volatile("s_wait_asynccnt 0x0" ::: "memory");
    }
    __syncthreads();                                     // all waves' copies visible

    const char* ldsA = smem + (i & 1) * STAGE_BYTES;
    const char* ldsB = ldsA + SLAB_BYTES;

    bf16x16 af[2];
    #pragma unroll
    for (int a = 0; a < 2; ++a) {
      const char* pa = ldsA + (wm * 32 + a * 16 + l15) * ROW_STRIDE + kh * 16;
      bf16x8 lo = *(const bf16x8*)pa;
      bf16x8 hi = *(const bf16x8*)(pa + 32);
      af[a] = SHUF16(lo, hi);
    }
    #pragma unroll
    for (int j = 0; j < 4; ++j) {
      const char* pb = ldsB + (wn * 64 + j * 16 + l15) * ROW_STRIDE + kh * 16;
      bf16x8 lo = *(const bf16x8*)pb;
      bf16x8 hi = *(const bf16x8*)(pb + 32);
      bf16x16 bfr = SHUF16(lo, hi);
      acc[0][j] = wmma_bf16(af[0], bfr, acc[0][j]);
      acc[1][j] = wmma_bf16(af[1], bfr, acc[1][j]);
    }
    __syncthreads();                                     // done reading stage i
  }

  // C/D layout: element r -> row = base + r + 8*kh, col = base + (lane&15)
  #pragma unroll
  for (int a = 0; a < 2; ++a) {
    #pragma unroll
    for (int j = 0; j < 4; ++j) {
      const int col = bn0 + wn * 64 + j * 16 + l15;
      #pragma unroll
      for (int r = 0; r < 8; ++r) {
        const int row = bm0 + wm * 32 + a * 16 + r + kh * 8;
        const size_t idx = (size_t)row * N + col;
        const float v = acc[a][j][r];
        switch (mode) {
          case 0: outF[idx] = v; break;
          case 1: outF[idx] = v / (1.f + __expf(-v)); break;                 // silu
          case 2: outF[idx] = v + res[idx]; break;                           // +residual
          case 3: { float t = v > 0.f ? v : 0.f; outB[idx] = (bf16)(t * t); } break;
          case 4: outF[idx] = 1.f / (1.f + __expf(-v)); break;               // sigmoid
          case 5: outF[idx] += scale[idx] * v; break;                        // gated accum
        }
      }
    }
  }
}

// ---------------------------------------------------------------------------
// WKV5 scan: one block per (b,h). 256 threads; thread (quad,j) owns state
// rows i in [quad*16, quad*16+16) for column j. State lives in registers.
// ---------------------------------------------------------------------------
__global__ void wkv5_kernel(const float* __restrict__ r, const float* __restrict__ k,
                            const float* __restrict__ v, const float* __restrict__ w,
                            const float* __restrict__ u, float* __restrict__ att,
                            int T, int H) {
  const int bh = blockIdx.x;
  const int b  = bh / H;
  const int h  = bh % H;
  const int tid  = threadIdx.x;
  const int j    = tid & 63;
  const int quad = tid >> 6;            // 0..3

  __shared__ float rS[64], kS[64], vS[64];
  __shared__ float yS[4][64];

  float s[16], uu[16], dd[16];
  #pragma unroll
  for (int ii = 0; ii < 16; ++ii) {
    const int i = quad * 16 + ii;
    s[ii]  = 0.f;
    dd[ii] = __expf(-__expf(w[h * 64 + i]));
    uu[ii] = u[h * 64 + i];
  }

  const size_t strideT = (size_t)H * 64;  // == C
  const float* rp = r   + (size_t)b * T * strideT + h * 64;
  const float* kp = k   + (size_t)b * T * strideT + h * 64;
  const float* vp = v   + (size_t)b * T * strideT + h * 64;
  float*       ap = att + (size_t)b * T * strideT + h * 64;

  for (int t = 0; t < T; ++t) {
    if (tid < 64) { rS[tid] = rp[tid]; kS[tid] = kp[tid]; vS[tid] = vp[tid]; }
    __syncthreads();
    const float vj = vS[j];
    float y = 0.f;
    #pragma unroll
    for (int ii = 0; ii < 16; ++ii) {
      const int i = quad * 16 + ii;
      const float kv = kS[i] * vj;
      y += rS[i] * (uu[ii] * kv + s[ii]);
      s[ii] = dd[ii] * s[ii] + kv;
    }
    yS[quad][j] = y;
    __syncthreads();
    if (quad == 0)
      ap[j] = yS[0][j] + yS[1][j] + yS[2][j] + yS[3][j];
    __syncthreads();   // protect rS/kS/vS/yS before next step's writes
    rp += strideT; kp += strideT; vp += strideT; ap += strideT;
  }
}

// ---------------------------------------------------------------------------
// GroupNorm(att/8) * lnx_g + lnx_b, gated by g (silu already applied) -> bf16
// ---------------------------------------------------------------------------
__global__ void gn_gate_kernel(const float* __restrict__ att, const float* __restrict__ gate,
                               const float* __restrict__ gw, const float* __restrict__ gb,
                               bf16* __restrict__ out, int C) {
  __shared__ float s1[256], s2[256];
  const int tid = threadIdx.x;
  const int h = tid >> 3;               // head 0..31
  const int l = tid & 7;                // 8 threads per head
  const size_t base = (size_t)blockIdx.x * C + h * 64;
  float a[8];
  float sum = 0.f, sum2 = 0.f;
  #pragma unroll
  for (int i = 0; i < 8; ++i) {
    const float v = att[base + l + 8 * i] * INV_DIV;
    a[i] = v; sum += v; sum2 += v * v;
  }
  s1[tid] = sum; s2[tid] = sum2;
  __syncthreads();
  for (int off = 4; off >= 1; off >>= 1) {
    if (l < off) { s1[tid] += s1[tid + off]; s2[tid] += s2[tid + off]; }
    __syncthreads();
  }
  const float mean = s1[h * 8] * (1.f / 64.f);
  const float var  = s2[h * 8] * (1.f / 64.f) - mean * mean;
  const float rstd = rsqrtf(var + EPSf);
  #pragma unroll
  for (int i = 0; i < 8; ++i) {
    const int c = h * 64 + l + 8 * i;
    const size_t idx = (size_t)blockIdx.x * C + c;
    const float xn = (a[i] - mean) * rstd;
    out[idx] = (bf16)((xn * gw[c] + gb[c]) * gate[idx]);
  }
}

// ---------------------------------------------------------------------------
// Host driver
// ---------------------------------------------------------------------------
extern "C" void kernel_launch(void* const* d_in, const int* in_sizes, int n_in,
                              void* d_out, int out_size, void* d_ws, size_t ws_size,
                              hipStream_t stream) {
  const float* x     = (const float*)d_in[0];
  const float* ln1_g = (const float*)d_in[1];
  const float* ln1_b = (const float*)d_in[2];
  const float* ln2_g = (const float*)d_in[3];
  const float* ln2_b = (const float*)d_in[4];
  const float* tm_k  = (const float*)d_in[5];
  const float* tm_v  = (const float*)d_in[6];
  const float* tm_r  = (const float*)d_in[7];
  const float* tm_g  = (const float*)d_in[8];
  const float* tdec  = (const float*)d_in[9];
  const float* tfaa  = (const float*)d_in[10];
  const float* Wr    = (const float*)d_in[11];
  const float* Wk    = (const float*)d_in[12];
  const float* Wv    = (const float*)d_in[13];
  const float* Wg    = (const float*)d_in[14];
  const float* Wo    = (const float*)d_in[15];
  const float* lnx_g = (const float*)d_in[16];
  const float* lnx_b = (const float*)d_in[17];
  const float* fm_k  = (const float*)d_in[18];
  const float* fm_r  = (const float*)d_in[19];
  const float* Wkey  = (const float*)d_in[20];
  const float* Wrec  = (const float*)d_in[21];
  const float* Wval  = (const float*)d_in[22];
  float* out = (float*)d_out;

  char* ws = (char*)d_ws;
  size_t off = 0;
  auto take = [&](size_t bytes) -> char* {
    char* p = ws + off;
    off += (bytes + 255) & ~(size_t)255;
    return p;
  };

  // bf16 transposed weights (112 MB total: resident in 192MB L2 across GEMMs)
  bf16* WrT   = (bf16*)take((size_t)Cc * Cc  * 2);
  bf16* WkT   = (bf16*)take((size_t)Cc * Cc  * 2);
  bf16* WvT   = (bf16*)take((size_t)Cc * Cc  * 2);
  bf16* WgT   = (bf16*)take((size_t)Cc * Cc  * 2);
  bf16* WoT   = (bf16*)take((size_t)Cc * Cc  * 2);
  bf16* WrecT = (bf16*)take((size_t)Cc * Cc  * 2);
  bf16* WkeyT = (bf16*)take((size_t)Cc * FFc * 2);  // [FF][C]
  bf16* WvalT = (bf16*)take((size_t)FFc * Cc * 2);  // [C][FF]
  // activations
  float* xl   = (float*)take(BTC * 4);              // reused for xl2
  bf16* xrb   = (bf16*)take(BTC * 2);
  bf16* xkb   = (bf16*)take(BTC * 2);
  bf16* xvb   = (bf16*)take(BTC * 2);
  bf16* xgb   = (bf16*)take(BTC * 2);
  float* rbuf = (float*)take(BTC * 4);
  float* kbuf = (float*)take(BTC * 4);
  float* vbuf = (float*)take(BTC * 4);
  float* gbuf = (float*)take(BTC * 4);
  float* attb = (float*)take(BTC * 4);
  bf16* gng   = (bf16*)take(BTC * 2);
  // aliases (lifetimes disjoint): sig over rbuf; ck/cr over k/v; kk over xr..xg
  float* sig  = rbuf;
  bf16*  ck   = (bf16*)kbuf;
  bf16*  cr   = (bf16*)vbuf;
  bf16*  kk   = xrb;            // spans xr..xg (32 MB, contiguous allocations)

  const dim3 tb(32, 8);
  transpose_to_bf16<<<dim3(Cc / 32,  Cc / 32),  tb, 0, stream>>>(Wr,   WrT,   Cc,  Cc);
  transpose_to_bf16<<<dim3(Cc / 32,  Cc / 32),  tb, 0, stream>>>(Wk,   WkT,   Cc,  Cc);
  transpose_to_bf16<<<dim3(Cc / 32,  Cc / 32),  tb, 0, stream>>>(Wv,   WvT,   Cc,  Cc);
  transpose_to_bf16<<<dim3(Cc / 32,  Cc / 32),  tb, 0, stream>>>(Wg,   WgT,   Cc,  Cc);
  transpose_to_bf16<<<dim3(Cc / 32,  Cc / 32),  tb, 0, stream>>>(Wo,   WoT,   Cc,  Cc);
  transpose_to_bf16<<<dim3(Cc / 32,  Cc / 32),  tb, 0, stream>>>(Wrec, WrecT, Cc,  Cc);
  transpose_to_bf16<<<dim3(FFc / 32, Cc / 32),  tb, 0, stream>>>(Wkey, WkeyT, Cc,  FFc);
  transpose_to_bf16<<<dim3(Cc / 32,  FFc / 32), tb, 0, stream>>>(Wval, WvalT, FFc, Cc);

  // ln1 + time-shift mixes
  layernorm_kernel<<<BT, 256, 0, stream>>>(x, ln1_g, ln1_b, xl, Cc);
  mix4_kernel<<<(int)(BTC / 256), 256, 0, stream>>>(xl, tm_k, tm_v, tm_r, tm_g,
                                                    xkb, xvb, xrb, xgb, Tc, Cc);

  // r/k/v/g projections (block tile 128x128)
  const dim3 gC(Cc / 128, BT / 128);    // (16, 16)
  gemm_bf16<<<gC, 256, 0, stream>>>(xrb, WrT, rbuf, nullptr, nullptr, nullptr, BT, Cc, Cc, 0);
  gemm_bf16<<<gC, 256, 0, stream>>>(xkb, WkT, kbuf, nullptr, nullptr, nullptr, BT, Cc, Cc, 0);
  gemm_bf16<<<gC, 256, 0, stream>>>(xvb, WvT, vbuf, nullptr, nullptr, nullptr, BT, Cc, Cc, 0);
  gemm_bf16<<<gC, 256, 0, stream>>>(xgb, WgT, gbuf, nullptr, nullptr, nullptr, BT, Cc, Cc, 1);

  // sequential WKV5 scan
  wkv5_kernel<<<Bc * Hc, 256, 0, stream>>>(rbuf, kbuf, vbuf, tdec, tfaa, attb, Tc, Hc);

  // groupnorm(att/8) * gate -> bf16, then Wo projection + residual into d_out
  gn_gate_kernel<<<BT, 256, 0, stream>>>(attb, gbuf, lnx_g, lnx_b, gng, Cc);
  gemm_bf16<<<gC, 256, 0, stream>>>(gng, WoT, out, nullptr, x, nullptr, BT, Cc, Cc, 2);

  // channel-mix half
  layernorm_kernel<<<BT, 256, 0, stream>>>(out, ln2_g, ln2_b, xl, Cc);
  mix2_kernel<<<(int)(BTC / 256), 256, 0, stream>>>(xl, fm_k, fm_r, ck, cr, Tc, Cc);

  const dim3 gFF(FFc / 128, BT / 128);  // (64, 16)
  gemm_bf16<<<gFF, 256, 0, stream>>>(ck, WkeyT, nullptr, kk, nullptr, nullptr, BT, FFc, Cc, 3);
  gemm_bf16<<<gC,  256, 0, stream>>>(cr, WrecT, sig, nullptr, nullptr, nullptr, BT, Cc, Cc, 4);
  gemm_bf16<<<gC,  256, 0, stream>>>(kk, WvalT, out, nullptr, nullptr, sig, BT, Cc, FFc, 5);

  (void)in_sizes; (void)n_in; (void)out_size; (void)ws_size;
}
